// FraudGNN_31645319037563
// MI455X (gfx1250) — compile-verified, matrix operand order checked by reading, not run on hardware
//
#include <hip/hip_runtime.h>
#include <hip/hip_bf16.h>

typedef __attribute__((ext_vector_type(16))) _Float16 v16h;
typedef __attribute__((ext_vector_type(8)))  _Float16 v8h;
typedef __attribute__((ext_vector_type(8)))  float    v8f;

#define NNODES 100000
#define NTILES (NNODES / 16)   // 6250, exact
#define WAVES_PER_BLOCK 8

// ---------------------------------------------------------------- zero scratch
__global__ void k_zero(float* __restrict__ p, long n) {
    long i = (long)blockIdx.x * blockDim.x + threadIdx.x;
    long stride = (long)gridDim.x * blockDim.x;
    for (; i < n; i += stride) p[i] = 0.0f;
}

// ---------------------------------------------------------------- degree count
__global__ void k_count(const int* __restrict__ dst, float* __restrict__ cnt, int E) {
    int e = blockIdx.x * blockDim.x + threadIdx.x;
    if (e < E) atomicAdd(&cnt[dst[e]], 1.0f);
}

// ------------------------------------------------ edge scatter (D % 4 == 0)
// D/4 threads per edge: coalesced float4 gather of feat[src], f32 atomics into agg[dst]
template <int D>
__global__ void k_scatter(const float* __restrict__ feat, const int* __restrict__ src,
                          const int* __restrict__ dst, float* __restrict__ agg, int E) {
    const int Q = D / 4;
    long tid = (long)blockIdx.x * blockDim.x + threadIdx.x;
    if (tid >= (long)E * Q) return;
    int e = (int)(tid / Q);
    int q = (int)(tid % Q);
    int s = src[e], d = dst[e];
    float4 v = ((const float4*)(feat + (long)s * D))[q];
    float* out = agg + (long)d * D + q * 4;
    atomicAdd(out + 0, v.x);
    atomicAdd(out + 1, v.y);
    atomicAdd(out + 2, v.z);
    atomicAdd(out + 3, v.w);
}

// stitch two 8-half groups into one 16-half WMMA fragment
__device__ __forceinline__ v16h join16(v8h lo, v8h hi) {
    return __builtin_shufflevector(lo, hi, 0, 1, 2, 3, 4, 5, 6, 7,
                                   8, 9, 10, 11, 12, 13, 14, 15);
}

// ------------------------------------------------ layer 1: h1 = relu(mean@W1l + b1 + x@W1r)
// [100k x 48] @ [48 x 32], K padded 48->64, per-wave 16x32 tile, 8 WMMAs/tile
// Weights staged in LDS TRANSPOSED [N][K] so each B half-group is one ds_load_b128.
__global__ void __launch_bounds__(256) k_sage1(
    const float* __restrict__ x, const float* __restrict__ agg1,
    const float* __restrict__ cnt, const float* __restrict__ W1l,
    const float* __restrict__ W1r, const float* __restrict__ b1,
    float* __restrict__ h1) {
    __shared__ _Float16 sWl[32][64];   // [n][k], K padded to 64
    __shared__ _Float16 sWr[32][64];
    for (int i = threadIdx.x; i < 32 * 64; i += 256) {
        int n = i >> 6, k = i & 63;
        sWl[n][k] = (k < 48) ? (_Float16)W1l[k * 32 + n] : (_Float16)0.0f;
        sWr[n][k] = (k < 48) ? (_Float16)W1r[k * 32 + n] : (_Float16)0.0f;
    }
    __syncthreads();

    const int lane = threadIdx.x & 31;
    const int wave = threadIdx.x >> 5;
    const int tile = blockIdx.x * WAVES_PER_BLOCK + wave;
    if (tile >= NTILES) return;                       // wave-uniform: EXEC stays all-1s

    const int hi16  = (lane >= 16) ? 1 : 0;
    const int kbase = hi16 ? 8 : 0;                   // ISA 16-bit A/B lane K-offset
    const int col   = lane & 15;
    const int row   = tile * 16 + col;                // A-matrix: lane -> row M

    const float inv = 1.0f / fmaxf(cnt[row], 1.0f);
    const float* xr = x    + (long)row * 48;
    const float* gr = agg1 + (long)row * 48;

    // A fragments: aX = x row, aM = mean row; chunk c covers K = c*32 .. c*32+31
    v16h aM[2], aX[2];
#pragma unroll
    for (int c = 0; c < 2; ++c)
#pragma unroll
        for (int g = 0; g < 2; ++g)
#pragma unroll
            for (int j = 0; j < 8; ++j) {
                int K = c * 32 + g * 16 + kbase + j;
                _Float16 vx = (K < 48) ? (_Float16)xr[K]         : (_Float16)0.0f;
                _Float16 vm = (K < 48) ? (_Float16)(gr[K] * inv) : (_Float16)0.0f;
                aX[c][g * 8 + j] = vx;
                aM[c][g * 8 + j] = vm;
            }

    // B fragments from LDS: 2x ds_load_b128 per fragment ([n][k] layout, 16B aligned)
    v16h bL[2][2], bR[2][2];
#pragma unroll
    for (int nh = 0; nh < 2; ++nh) {
        int n = nh * 16 + col;
        const v8h* pl = (const v8h*)&sWl[n][0];       // 8 groups of 8 halves per row
        const v8h* pr = (const v8h*)&sWr[n][0];
#pragma unroll
        for (int c = 0; c < 2; ++c) {
            bL[nh][c] = join16(pl[c * 4 + hi16], pl[c * 4 + 2 + hi16]);
            bR[nh][c] = join16(pr[c * 4 + hi16], pr[c * 4 + 2 + hi16]);
        }
    }

#pragma unroll
    for (int nh = 0; nh < 2; ++nh) {
        int n = nh * 16 + col;
        float bias = b1[n];
        v8f acc;
#pragma unroll
        for (int r = 0; r < 8; ++r) acc[r] = bias;
#pragma unroll
        for (int c = 0; c < 2; ++c) {
            acc = __builtin_amdgcn_wmma_f32_16x16x32_f16(false, aM[c], false, bL[nh][c],
                                                         (short)0, acc, false, false);
            acc = __builtin_amdgcn_wmma_f32_16x16x32_f16(false, aX[c], false, bR[nh][c],
                                                         (short)0, acc, false, false);
        }
        int rbase = tile * 16 + hi16 * 8;             // C/D layout: lane>=16 -> rows M+8
#pragma unroll
        for (int r = 0; r < 8; ++r)
            h1[(long)(rbase + r) * 32 + n] = fmaxf(acc[r], 0.0f);
    }
}

// ------------------------------------------------ layer 2: h2 = relu(mean@W2l + b2 + h1@W2r)
// [100k x 32] @ [32 x 16], exact K=32, 2 WMMAs/tile
__global__ void __launch_bounds__(256) k_sage2(
    const float* __restrict__ h1, const float* __restrict__ agg2,
    const float* __restrict__ cnt, const float* __restrict__ W2l,
    const float* __restrict__ W2r, const float* __restrict__ b2,
    float* __restrict__ h2) {
    __shared__ _Float16 sWl[16][32];   // [n][k]
    __shared__ _Float16 sWr[16][32];
    for (int i = threadIdx.x; i < 16 * 32; i += 256) {
        int n = i >> 5, k = i & 31;
        sWl[n][k] = (_Float16)W2l[k * 16 + n];
        sWr[n][k] = (_Float16)W2r[k * 16 + n];
    }
    __syncthreads();

    const int lane = threadIdx.x & 31;
    const int wave = threadIdx.x >> 5;
    const int tile = blockIdx.x * WAVES_PER_BLOCK + wave;
    if (tile >= NTILES) return;

    const int hi16  = (lane >= 16) ? 1 : 0;
    const int kbase = hi16 ? 8 : 0;
    const int col   = lane & 15;
    const int row   = tile * 16 + col;

    const float inv = 1.0f / fmaxf(cnt[row], 1.0f);
    const float* hr = h1   + (long)row * 32;
    const float* gr = agg2 + (long)row * 32;

    v16h aM, aX;
#pragma unroll
    for (int g = 0; g < 2; ++g)
#pragma unroll
        for (int j = 0; j < 8; ++j) {
            int K = g * 16 + kbase + j;               // K in [0,32)
            aX[g * 8 + j] = (_Float16)hr[K];
            aM[g * 8 + j] = (_Float16)(gr[K] * inv);
        }

    const v8h* pl = (const v8h*)&sWl[col][0];         // 4 groups of 8 halves per row
    const v8h* pr = (const v8h*)&sWr[col][0];
    v16h bL = join16(pl[hi16], pl[2 + hi16]);
    v16h bR = join16(pr[hi16], pr[2 + hi16]);

    float bias = b2[col];
    v8f acc;
#pragma unroll
    for (int r = 0; r < 8; ++r) acc[r] = bias;
    acc = __builtin_amdgcn_wmma_f32_16x16x32_f16(false, aM, false, bL, (short)0, acc, false, false);
    acc = __builtin_amdgcn_wmma_f32_16x16x32_f16(false, aX, false, bR, (short)0, acc, false, false);

    int rbase = tile * 16 + hi16 * 8;
#pragma unroll
    for (int r = 0; r < 8; ++r)
        h2[(long)(rbase + r) * 16 + col] = fmaxf(acc[r], 0.0f);
}

// ------------------------------------------------ head: out = h2 @ Wlin + blin  (N=2 -> VALU)
__global__ void k_head(const float* __restrict__ h2, const float* __restrict__ Wlin,
                       const float* __restrict__ blin, float* __restrict__ out, int N) {
    int i = blockIdx.x * blockDim.x + threadIdx.x;
    if (i >= N) return;
    const float* h = h2 + (long)i * 16;
    float o0 = blin[0], o1 = blin[1];
#pragma unroll
    for (int k = 0; k < 16; ++k) {
        float v = h[k];
        o0 += v * Wlin[k * 2 + 0];
        o1 += v * Wlin[k * 2 + 1];
    }
    out[(long)i * 2 + 0] = o0;
    out[(long)i * 2 + 1] = o1;
}

// ---------------------------------------------------------------- launch
extern "C" void kernel_launch(void* const* d_in, const int* in_sizes, int n_in,
                              void* d_out, int out_size, void* d_ws, size_t ws_size,
                              hipStream_t stream) {
    const float* x    = (const float*)d_in[0];
    const int*   ei   = (const int*)d_in[1];          // [2, E] row-major (int32)
    const float* W1l  = (const float*)d_in[2];
    const float* W1r  = (const float*)d_in[3];
    const float* b1   = (const float*)d_in[4];
    const float* W2l  = (const float*)d_in[5];
    const float* W2r  = (const float*)d_in[6];
    const float* b2   = (const float*)d_in[7];
    const float* Wlin = (const float*)d_in[8];
    const float* blin = (const float*)d_in[9];

    const int E = in_sizes[1] / 2;
    const int* src = ei;
    const int* dst = ei + E;

    // workspace layout (floats): [agg1 | agg2 | cnt | h1 | h2]
    float* ws   = (float*)d_ws;
    float* agg1 = ws;                                  // NNODES*48
    float* agg2 = agg1 + (long)NNODES * 48;            // NNODES*32
    float* cnt  = agg2 + (long)NNODES * 32;            // NNODES
    float* h1   = cnt  + (long)NNODES;                 // NNODES*32
    float* h2   = h1   + (long)NNODES * 32;            // NNODES*16

    const long zeroN = (long)NNODES * (48 + 32 + 1);   // agg1 + agg2 + cnt
    k_zero<<<2048, 256, 0, stream>>>(ws, zeroN);

    k_count<<<(E + 255) / 256, 256, 0, stream>>>(dst, cnt, E);

    {   // scatter layer-1 features: 12 float4 lanes per edge
        long t = (long)E * 12;
        k_scatter<48><<<(unsigned)((t + 255) / 256), 256, 0, stream>>>(x, src, dst, agg1, E);
    }

    {   // layer 1 WMMA
        int blocks = (NTILES + WAVES_PER_BLOCK - 1) / WAVES_PER_BLOCK;
        k_sage1<<<blocks, 256, 0, stream>>>(x, agg1, cnt, W1l, W1r, b1, h1);
    }

    {   // scatter layer-2 features: 8 float4 lanes per edge
        long t = (long)E * 8;
        k_scatter<32><<<(unsigned)((t + 255) / 256), 256, 0, stream>>>(h1, src, dst, agg2, E);
    }

    {   // layer 2 WMMA
        int blocks = (NTILES + WAVES_PER_BLOCK - 1) / WAVES_PER_BLOCK;
        k_sage2<<<blocks, 256, 0, stream>>>(h1, agg2, cnt, W2l, W2r, b2, h2);
    }

    k_head<<<(NNODES + 255) / 256, 256, 0, stream>>>(h2, Wlin, blin, (float*)d_out, NNODES);
}
